// LSTM_Attn_2353642078928
// MI455X (gfx1250) — matrix operand order repaired and checked by reading, not compiled, not run on hardware
//
#include <hip/hip_runtime.h>
#include <hip/hip_bf16.h>
#include <math.h>

// ---------------------------------------------------------------------------
// Problem geometry (fixed by the reference)
// ---------------------------------------------------------------------------
#define CCH   64        // channel_size
#define ATT   128       // att_size
#define HID   256       // hidden_size
#define NACT  4         // num_actions
#define HWDIM 512
#define NPOS  (HWDIM*HWDIM)     // 262144 spatial positions
#define POS_PER_BLOCK 128       // 8 waves * 16 positions
#define NBLK  (NPOS/POS_PER_BLOCK)  // 2048
#define NFRAG ((NPOS/16)*2)     // img A-fragments: 16-pos tile x 2 K-blocks

typedef __attribute__((ext_vector_type(16))) __bf16 v16bf;
typedef __attribute__((ext_vector_type(8)))  float  v8f;

// ---------------------------------------------------------------------------
// helpers
// ---------------------------------------------------------------------------
__device__ __forceinline__ unsigned short f2bf(float f) {
    // native f32 -> bf16 (hardware cvt, RNE)
    return __builtin_bit_cast(unsigned short, (__bf16)f);
}
__device__ __forceinline__ __bf16 bfbits(unsigned short u) {
    return __builtin_bit_cast(__bf16, u);
}
// monotone float<->u32 encoding so atomicMax(u32) == float max (exact, order-free)
__device__ __forceinline__ unsigned fenc(float f) {
    unsigned u = __float_as_uint(f);
    return (u & 0x80000000u) ? ~u : (u | 0x80000000u);
}
__device__ __forceinline__ float fdec(unsigned u) {
    return __uint_as_float((u & 0x80000000u) ? (u ^ 0x80000000u) : ~u);
}
__device__ __forceinline__ float sigmoidf(float x) { return 1.0f / (1.0f + expf(-x)); }

// A-fragment K mapping for 16-bit 16x32 A (ISA 7.12.2): granules of 8 K per half-wave
__device__ __forceinline__ int kmapA(int e, int half) {
    return (e < 8) ? (8 * half + e) : (8 + 8 * half + e);
}
// B-fragment K mapping for 16-bit 32x16 B: same granule-of-8 order as A
__device__ __forceinline__ int kmapB(int e, int half) {
    return (e < 8) ? (8 * half + e) : (8 + 8 * half + e);
}

// A fragment from img, column-major over K (img layout: [C][N]); f32 fallback
__device__ __forceinline__ v16bf load_afrag_img(const float* __restrict__ img,
                                                int pos0, int c0) {
    int lane = threadIdx.x & 31, m = lane & 15, half = lane >> 4;
    v16bf a;
#pragma unroll
    for (int e = 0; e < 16; ++e) {
        int k = c0 + kmapA(e, half);
        a[e] = (__bf16)img[(size_t)k * NPOS + pos0 + m];
    }
    return a;
}
// Fragment-native tiled bf16 source: one aligned 32-byte vector load per lane
__device__ __forceinline__ v16bf load_afrag_img(const unsigned short* __restrict__ img,
                                                int pos0, int c0) {
    int lane = threadIdx.x & 31;
    size_t frag = ((size_t)(pos0 >> 4) << 1) + (size_t)(c0 >> 5);
    return *reinterpret_cast<const v16bf*>(img + (frag * 32 + lane) * 16);
}
// A fragment from LDS bf16 feat tile, row-major [m][128]
__device__ __forceinline__ v16bf load_afrag_lds(const unsigned short* __restrict__ f,
                                                int k0) {
    int lane = threadIdx.x & 31, m = lane & 15, half = lane >> 4;
    v16bf a;
#pragma unroll
    for (int e = 0; e < 16; ++e)
        a[e] = bfbits(f[m * ATT + k0 + kmapA(e, half)]);
    return a;
}
// B fragment from bf16 weights stored as Bt[n][k] row-major (i.e. W[a][k] as given)
__device__ __forceinline__ v16bf load_bfrag(const unsigned short* __restrict__ w,
                                            int n0, int k0, int ldk) {
    int lane = threadIdx.x & 31, nl = lane & 15, half = lane >> 4;
    v16bf b;
#pragma unroll
    for (int e = 0; e < 16; ++e)
        b[e] = bfbits(w[(n0 + nl) * ldk + k0 + kmapB(e, half)]);
    return b;
}

// conv tile: feat[16 x 128] = relu(imgT[16 x 64] * conv_wT[64 x 128] + b), bf16 WMMA
template <typename IMGT>
__device__ __forceinline__ void conv_feat(const IMGT* __restrict__ img,
                                          const unsigned short* __restrict__ cwbf,
                                          const float* __restrict__ conv_b,
                                          int pos0, v8f acc[8]) {
    const v8f vzero = {0.f, 0.f, 0.f, 0.f, 0.f, 0.f, 0.f, 0.f};
#pragma unroll
    for (int t = 0; t < 8; ++t) acc[t] = vzero;
#pragma unroll
    for (int kk = 0; kk < CCH; kk += 32) {
        v16bf af = load_afrag_img(img, pos0, kk);
#pragma unroll
        for (int t = 0; t < 8; ++t) {
            v16bf bf = load_bfrag(cwbf, t * 16, kk, CCH);
            acc[t] = __builtin_amdgcn_wmma_f32_16x16x32_bf16(
                false, af, false, bf, (short)0, acc[t], false, false);
        }
    }
    int nl = threadIdx.x & 15;
#pragma unroll
    for (int t = 0; t < 8; ++t) {
        float bias = conv_b[t * 16 + nl];
#pragma unroll
        for (int r = 0; r < 8; ++r) {
            float v = acc[t][r] + bias;
            acc[t][r] = v > 0.f ? v : 0.f;
        }
    }
}

// ---------------------------------------------------------------------------
// Kernel 0: repack img_code f32 -> fragment-native tiled bf16
// thread <-> (frag, lane); reads coalesced f32, writes one 32B chunk
// ---------------------------------------------------------------------------
__global__ __launch_bounds__(256) void k_img2tiled(const float* __restrict__ img,
                                                   unsigned short* __restrict__ out) {
    unsigned idx = blockIdx.x * 256 + threadIdx.x;
    unsigned frag = idx >> 5;
    int lane = idx & 31;
    int m = lane & 15, half = (lane >> 4) & 1;
    int pos0 = (int)((frag >> 1) << 4);
    int kk = (int)((frag & 1) << 5);
    v16bf a;
#pragma unroll
    for (int e = 0; e < 16; ++e) {
        int k = kk + kmapA(e, half);
        a[e] = (__bf16)img[(size_t)k * NPOS + pos0 + m];
    }
    *reinterpret_cast<v16bf*>(out + ((size_t)frag * 32 + lane) * 16) = a;
}

// ---------------------------------------------------------------------------
// Kernel 1: LSTM step + att2 + weight conversion + init (1 block, 256 thr)
// ---------------------------------------------------------------------------
__global__ __launch_bounds__(256) void k_setup(
    const float* __restrict__ action, const float* __restrict__ h0,
    const float* __restrict__ c0,
    const float* __restrict__ ae_w, const float* __restrict__ ae_b,
    const float* __restrict__ w_ih, const float* __restrict__ w_hh,
    const float* __restrict__ b_ih, const float* __restrict__ b_hh,
    const float* __restrict__ ha_w, const float* __restrict__ ha_b,
    const float* __restrict__ fa_w, const float* __restrict__ fa_b,
    const float* __restrict__ conv_w, const float* __restrict__ ia_w,
    unsigned short* __restrict__ cwbf, unsigned short* __restrict__ iawbf,
    float* __restrict__ h_ws, float* __restrict__ sconst_p,
    unsigned* __restrict__ gmax_enc, float* __restrict__ d_out) {
    __shared__ float xs[HID], h0s[HID], hsh[HID], a2[ATT];
    int tid = threadIdx.x;

    // weight f32 -> bf16 conversion into workspace
    for (int i = tid; i < ATT * CCH; i += 256) cwbf[i] = f2bf(conv_w[i]);
    for (int i = tid; i < ATT * ATT; i += 256) iawbf[i] = f2bf(ia_w[i]);

    // action encoder: x = action @ ae_w.T + ae_b
    float x = ae_b[tid];
#pragma unroll
    for (int k = 0; k < NACT; ++k) x += action[k] * ae_w[tid * NACT + k];
    xs[tid] = x;
    h0s[tid] = h0[tid];
    __syncthreads();

    // LSTM gates (PyTorch order i,f,g,o), one cell per thread
    float g[4];
#pragma unroll
    for (int q = 0; q < 4; ++q) {
        int row = q * HID + tid;
        float acc = b_ih[row] + b_hh[row];
        const float* wi = w_ih + (size_t)row * HID;
        const float* wh = w_hh + (size_t)row * HID;
        for (int k = 0; k < HID; ++k) acc += xs[k] * wi[k] + h0s[k] * wh[k];
        g[q] = acc;
    }
    float c = sigmoidf(g[1]) * c0[tid] + sigmoidf(g[0]) * tanhf(g[2]);
    float h = sigmoidf(g[3]) * tanhf(c);
    d_out[4 + tid] = h;          // h output
    d_out[4 + HID + tid] = c;    // c output
    h_ws[tid] = h;
    hsh[tid] = h;
    __syncthreads();

    // att2 = relu(h @ ha_w.T + ha_b); only its fa_w-projection matters (softmax const)
    if (tid < ATT) {
        float a = ha_b[tid];
        for (int k = 0; k < HID; ++k) a += hsh[k] * ha_w[tid * HID + k];
        a2[tid] = a > 0.f ? a : 0.f;
    }
    __syncthreads();
    if (tid == 0) {
        float s = fa_b[0];
        for (int a = 0; a < ATT; ++a) s += fa_w[a] * a2[a];
        *sconst_p = s;
        *gmax_enc = 0x007FFFFFu;  // fenc(-inf)
    }
}

// ---------------------------------------------------------------------------
// Kernel 2: fused conv + att1 + score pass (WMMA), scores + global max
// ---------------------------------------------------------------------------
template <typename IMGT>
__global__ __launch_bounds__(256) void k_scores(
    const IMGT* __restrict__ img,
    const unsigned short* __restrict__ cwbf,
    const unsigned short* __restrict__ iawbf,
    const float* __restrict__ conv_b, const float* __restrict__ ia_b,
    const float* __restrict__ fa_w,
    const float* __restrict__ sconst_p,
    float* __restrict__ scores, unsigned* __restrict__ gmax_enc) {
    __shared__ unsigned short featS[8 * 16 * ATT];  // 32 KB: per-wave 16x128 bf16
    __shared__ float redS[16];

    int wave = threadIdx.x >> 5;
    int lane = threadIdx.x & 31;
    int nl = lane & 15, half = lane >> 4;
    int pos0 = blockIdx.x * POS_PER_BLOCK + wave * 16;

    // ---- conv tile via WMMA ----
    v8f acc[8];
    conv_feat(img, cwbf, conv_b, pos0, acc);

    // D-layout -> row-major bf16 tile in LDS (transpose via LDS)
    unsigned short* fS = featS + wave * 16 * ATT;
#pragma unroll
    for (int t = 0; t < 8; ++t)
#pragma unroll
        for (int r = 0; r < 8; ++r) {
            int m = r + 8 * half;
            fS[m * ATT + t * 16 + nl] = f2bf(acc[t][r]);
        }
    __syncthreads();

    // ---- att1 = relu(feat @ ia_w.T + ia_b) via WMMA, fold fa_w dot ----
    const v8f vzero = {0.f, 0.f, 0.f, 0.f, 0.f, 0.f, 0.f, 0.f};
    v8f a1[8];
#pragma unroll
    for (int t = 0; t < 8; ++t) a1[t] = vzero;
#pragma unroll
    for (int k = 0; k < ATT; k += 32) {
        v16bf af = load_afrag_lds(fS, k);
#pragma unroll
        for (int t = 0; t < 8; ++t) {
            v16bf bf = load_bfrag(iawbf, t * 16, k, ATT);
            a1[t] = __builtin_amdgcn_wmma_f32_16x16x32_bf16(
                false, af, false, bf, (short)0, a1[t], false, false);
        }
    }
    float sacc[8] = {0.f, 0.f, 0.f, 0.f, 0.f, 0.f, 0.f, 0.f};
#pragma unroll
    for (int t = 0; t < 8; ++t) {
        float bias = ia_b[t * 16 + nl];
        float fw = fa_w[t * 16 + nl];
#pragma unroll
        for (int r = 0; r < 8; ++r) {
            float v = a1[t][r] + bias;
            v = v > 0.f ? v : 0.f;
            sacc[r] += fw * v;
        }
    }
    // reduce across the 16 lanes of each half (each half covers 8 rows)
#pragma unroll
    for (int s = 1; s < 16; s <<= 1)
#pragma unroll
        for (int r = 0; r < 8; ++r) sacc[r] += __shfl_xor(sacc[r], s, 32);

    float sconst = *sconst_p;
    float lmax = -INFINITY;
    if (nl == 0) {
#pragma unroll
        for (int r = 0; r < 8; ++r) {
            float s = sacc[r] + sconst;
            scores[pos0 + 8 * half + r] = s;
            lmax = fmaxf(lmax, s);
        }
        redS[wave * 2 + half] = lmax;
    }
    __syncthreads();
    if (threadIdx.x == 0) {
        float m = -INFINITY;
#pragma unroll
        for (int i = 0; i < 16; ++i) m = fmaxf(m, redS[i]);
        atomicMax(gmax_enc, fenc(m));
    }
}

// ---------------------------------------------------------------------------
// Kernel 3: recompute conv tile, accumulate exp-weighted feature partials
// ---------------------------------------------------------------------------
template <typename IMGT>
__global__ __launch_bounds__(256) void k_weight(
    const IMGT* __restrict__ img,
    const unsigned short* __restrict__ cwbf,
    const float* __restrict__ conv_b,
    const float* __restrict__ scores,
    const unsigned* __restrict__ gmax_enc,
    float* __restrict__ wpart, float* __restrict__ ssum) {
    __shared__ float wS[8 * ATT];
    __shared__ float sS[8];

    int wave = threadIdx.x >> 5;
    int lane = threadIdx.x & 31;
    int nl = lane & 15, half = lane >> 4;
    int pos0 = blockIdx.x * POS_PER_BLOCK + wave * 16;

    float gmax = fdec(*gmax_enc);

    v8f acc[8];
    conv_feat(img, cwbf, conv_b, pos0, acc);

    float e[8], esum = 0.f;
#pragma unroll
    for (int r = 0; r < 8; ++r) {
        e[r] = expf(scores[pos0 + 8 * half + r] - gmax);
        esum += e[r];
    }
#pragma unroll
    for (int t = 0; t < 8; ++t) {
        float wa = 0.f;
#pragma unroll
        for (int r = 0; r < 8; ++r) wa += e[r] * acc[t][r];
        wa += __shfl_xor(wa, 16, 32);  // combine the two half-wave row sets
        if (half == 0) wS[wave * ATT + t * 16 + nl] = wa;
    }
    esum += __shfl_xor(esum, 16, 32);
    if (lane == 0) sS[wave] = esum;
    __syncthreads();

    int tid = threadIdx.x;
    if (tid < ATT) {  // fixed-order reduction across waves => deterministic
        float s = 0.f;
#pragma unroll
        for (int w = 0; w < 8; ++w) s += wS[w * ATT + tid];
        wpart[(size_t)blockIdx.x * ATT + tid] = s;
    }
    if (tid == ATT) {
        float s = 0.f;
#pragma unroll
        for (int w = 0; w < 8; ++w) s += sS[w];
        ssum[blockIdx.x] = s;
    }
}

// ---------------------------------------------------------------------------
// Kernel 4: final reduction + actor head (1 block)
// ---------------------------------------------------------------------------
__global__ __launch_bounds__(256) void k_final(
    const float* __restrict__ wpart, const float* __restrict__ ssum,
    const float* __restrict__ h_ws,
    const float* __restrict__ actor_w, const float* __restrict__ actor_b,
    float* __restrict__ d_out) {
    __shared__ float attS[ATT], red[128], Ssh;
    int tid = threadIdx.x;
    if (tid < ATT) {
        float s = 0.f;
        for (int b = 0; b < NBLK; ++b) s += wpart[(size_t)b * ATT + tid];
        attS[tid] = s;
    } else {
        int j = tid - 128;
        float s = 0.f;
        for (int b = j; b < NBLK; b += 128) s += ssum[b];
        red[j] = s;
    }
    __syncthreads();
    if (tid == 0) {
        float S = 0.f;
        for (int j = 0; j < 128; ++j) S += red[j];
        Ssh = S;
    }
    __syncthreads();
    if (tid < ATT) attS[tid] /= Ssh;
    __syncthreads();
    if (tid < NACT) {
        float acc = actor_b[tid];
        const float* aw = actor_w + tid * (ATT + HID);
        for (int a = 0; a < ATT; ++a) acc += aw[a] * attS[a];
        for (int k = 0; k < HID; ++k) acc += aw[ATT + k] * h_ws[k];
        d_out[tid] = acc;
    }
}

// ---------------------------------------------------------------------------
// launch
// ---------------------------------------------------------------------------
extern "C" void kernel_launch(void* const* d_in, const int* in_sizes, int n_in,
                              void* d_out, int out_size, void* d_ws, size_t ws_size,
                              hipStream_t stream) {
    const float* img     = (const float*)d_in[0];
    const float* action  = (const float*)d_in[1];
    const float* h0      = (const float*)d_in[2];
    const float* c0      = (const float*)d_in[3];
    const float* conv_w  = (const float*)d_in[4];
    const float* conv_b  = (const float*)d_in[5];
    const float* ae_w    = (const float*)d_in[6];
    const float* ae_b    = (const float*)d_in[7];
    const float* ia_w    = (const float*)d_in[8];
    const float* ia_b    = (const float*)d_in[9];
    const float* ha_w    = (const float*)d_in[10];
    const float* ha_b    = (const float*)d_in[11];
    const float* fa_w    = (const float*)d_in[12];
    const float* fa_b    = (const float*)d_in[13];
    const float* w_ih    = (const float*)d_in[14];
    const float* w_hh    = (const float*)d_in[15];
    const float* b_ih    = (const float*)d_in[16];
    const float* b_hh    = (const float*)d_in[17];
    const float* actor_w = (const float*)d_in[18];
    const float* actor_b = (const float*)d_in[19];
    float* out = (float*)d_out;

    // workspace carve (base ~2.1 MB; +32 MB for tiled bf16 image if it fits)
    char* w0 = (char*)d_ws;
    char* w = w0;
    unsigned short* cwbf  = (unsigned short*)w; w += (size_t)ATT * CCH * 2;
    unsigned short* iawbf = (unsigned short*)w; w += (size_t)ATT * ATT * 2;
    float* scores = (float*)w;                  w += (size_t)NPOS * 4;
    float* wpart  = (float*)w;                  w += (size_t)NBLK * ATT * 4;
    float* ssum   = (float*)w;                  w += (size_t)NBLK * 4;
    float* h_ws   = (float*)w;                  w += (size_t)HID * 4;
    float* sconst = (float*)w;                  w += 16;
    unsigned* gmax = (unsigned*)w;              w += 16;
    unsigned short* imgbf = (unsigned short*)w;  // 32B-aligned offset
    size_t need_bf = (size_t)(w - w0) + (size_t)NPOS * CCH * 2;
    bool use_bf = (ws_size >= need_bf);   // ws_size is fixed => deterministic

    k_setup<<<1, 256, 0, stream>>>(action, h0, c0, ae_w, ae_b, w_ih, w_hh,
                                   b_ih, b_hh, ha_w, ha_b, fa_w, fa_b,
                                   conv_w, ia_w, cwbf, iawbf, h_ws, sconst,
                                   gmax, out);
    if (use_bf) {
        // one-time repack into WMMA-fragment-native bf16 tiles: hot passes do
        // one aligned 32B vector load per A-fragment; 32 MB stays L2-resident.
        k_img2tiled<<<(NFRAG * 32) / 256, 256, 0, stream>>>(img, imgbf);
        k_scores<unsigned short><<<NBLK, 256, 0, stream>>>(
            imgbf, cwbf, iawbf, conv_b, ia_b, fa_w, sconst, scores, gmax);
        k_weight<unsigned short><<<NBLK, 256, 0, stream>>>(
            imgbf, cwbf, conv_b, scores, gmax, wpart, ssum);
    } else {
        k_scores<float><<<NBLK, 256, 0, stream>>>(
            img, cwbf, iawbf, conv_b, ia_b, fa_w, sconst, scores, gmax);
        k_weight<float><<<NBLK, 256, 0, stream>>>(
            img, cwbf, conv_b, scores, gmax, wpart, ssum);
    }
    k_final<<<1, 256, 0, stream>>>(wpart, ssum, h_ws, actor_w, actor_b, out);
}
